// RWKV_TimeMix_34136400068822
// MI455X (gfx1250) — matrix operand order repaired and checked by reading, not compiled, not run on hardware
//
#include <hip/hip_runtime.h>
#include <hip/hip_bf16.h>

#define TB 8
#define T_CTX 1024
#define C_EMBD 1024
#define NA 1024
#define NH 16
#define HS 64

#define BM 128
#define BN 128
#define BK 32

typedef __attribute__((ext_vector_type(16))) __bf16 v16bf;
typedef __attribute__((ext_vector_type(8)))  float  v8f;

union FragBF {
    uint4 q[2];
    v16bf v;
    unsigned short s[16];
};

__device__ __forceinline__ unsigned short f2bf(float f) {
    unsigned int u = __float_as_uint(f);
    u += 0x7FFFu + ((u >> 16) & 1u);   // round-to-nearest-even
    return (unsigned short)(u >> 16);
}

// CDNA5 async global->LDS copy (ASYNCcnt-tracked); vdst carries the LDS byte
// address (low 32 bits of the flat shared pointer == wave-relative LDS offset).
__device__ __forceinline__ void async_b128(const void* lds_ptr, const void* gptr) {
    unsigned lds = (unsigned)(size_t)lds_ptr;
    asm volatile("global_load_async_to_lds_b128 %0, %1, off"
                 :: "v"(lds), "v"(gptr)
                 : "memory");
}
__device__ __forceinline__ void wait_async0() {
    asm volatile("s_wait_asynccnt 0" ::: "memory");
}

// ---------------------------------------------------------------- prep kernels
__global__ __launch_bounds__(256)
void k_shift_cast(const float* __restrict__ x, unsigned short* __restrict__ xs, int total) {
    int i = blockIdx.x * blockDim.x + threadIdx.x;
    if (i >= total) return;
    int c = i % C_EMBD;
    int t = (i / C_EMBD) % T_CTX;
    float v;
    if (c < C_EMBD / 2) v = (t == 0) ? 0.0f : x[i - C_EMBD];
    else                v = x[i];
    xs[i] = f2bf(v);
}

__global__ __launch_bounds__(256)
void k_cast(const float* __restrict__ src, unsigned short* __restrict__ dst, int n) {
    int i = blockIdx.x * blockDim.x + threadIdx.x;
    if (i < n) dst[i] = f2bf(src[i]);
}

// ---------------------------------------------------------------- fused k/v/r GEMM
// P = xs @ W^T + bias, epilogue per z: 0 -> exp(clip), 1 -> identity, 2 -> sigmoid
__global__ __launch_bounds__(256)
void k_gemm_kvr(const unsigned short* __restrict__ A,
                const unsigned short* __restrict__ Wkb,
                const unsigned short* __restrict__ Wvb,
                const unsigned short* __restrict__ Wrb,
                const float* __restrict__ bk,
                const float* __restrict__ bv,
                const float* __restrict__ br,
                float* __restrict__ kexp,
                float* __restrict__ vout,
                float* __restrict__ sigr) {
    __shared__ unsigned short sA[2][BM][BK];
    __shared__ unsigned short sB[2][BN][BK];

    const int z = blockIdx.z;
    const unsigned short* W = (z == 0) ? Wkb : (z == 1) ? Wvb : Wrb;
    const int m0 = blockIdx.y * BM;
    const int n0 = blockIdx.x * BN;
    const int tid  = threadIdx.x;
    const int lane = tid & 31;
    const int wave = tid >> 5;
    const int wm = wave & 1;   // 2 waves over M (64 rows each)
    const int wn = wave >> 1;  // 4 waves over N (32 cols each)
    const int l15 = lane & 15;
    const int hi  = lane >> 4;

    v8f acc[4][2];
    #pragma unroll
    for (int i = 0; i < 4; ++i)
        #pragma unroll
        for (int j = 0; j < 2; ++j) acc[i][j] = 0.0f;

    const int arow = tid >> 1;          // 128 rows, 2 threads/row
    const int acol = (tid & 1) * 16;    // ushort offset

    auto stage = [&](int buf, int k0) {
        const unsigned short* ga = A + (size_t)(m0 + arow) * C_EMBD + k0 + acol;
        async_b128(&sA[buf][arow][acol],     ga);
        async_b128(&sA[buf][arow][acol + 8], ga + 8);
        const unsigned short* gb = W + (size_t)(n0 + arow) * C_EMBD + k0 + acol;
        async_b128(&sB[buf][arow][acol],     gb);
        async_b128(&sB[buf][arow][acol + 8], gb + 8);
    };

    auto compute = [&](int buf) {
        FragBF bfr[2];
        #pragma unroll
        for (int nt = 0; nt < 2; ++nt) {
            const int n = wn * 32 + nt * 16 + l15;
            bfr[nt].q[0] = *(const uint4*)&sB[buf][n][hi * 8];
            bfr[nt].q[1] = *(const uint4*)&sB[buf][n][16 + hi * 8];
        }
        #pragma unroll
        for (int mt = 0; mt < 4; ++mt) {
            FragBF a;
            const int m = wm * 64 + mt * 16 + l15;
            a.q[0] = *(const uint4*)&sA[buf][m][hi * 8];        // K slots 0..7
            a.q[1] = *(const uint4*)&sA[buf][m][16 + hi * 8];   // K slots 8..15
            #pragma unroll
            for (int nt = 0; nt < 2; ++nt)
                acc[mt][nt] = __builtin_amdgcn_wmma_f32_16x16x32_bf16(
                    false, a.v, false, bfr[nt].v, (short)0, acc[mt][nt], false, false);
        }
    };

    // branch-free double-buffered pipeline (nk == 32, even)
    const int nk = C_EMBD / BK;
    stage(0, 0);
    wait_async0();
    __syncthreads();

    #pragma unroll 1
    for (int i = 0; i < nk - 2; i += 2) {
        stage(1, (i + 1) * BK);
        compute(0);
        wait_async0();
        __syncthreads();
        stage(0, (i + 2) * BK);
        compute(1);
        wait_async0();
        __syncthreads();
    }
    stage(1, (nk - 1) * BK);
    compute(0);
    wait_async0();
    __syncthreads();
    compute(1);

    #pragma unroll
    for (int mt = 0; mt < 4; ++mt) {
        #pragma unroll
        for (int nt = 0; nt < 2; ++nt) {
            const int n = n0 + wn * 32 + nt * 16 + l15;
            const float bias = (z == 0) ? bk[n] : (z == 1) ? bv[n] : br[n];
            #pragma unroll
            for (int j = 0; j < 8; ++j) {
                const int m = m0 + wm * 64 + mt * 16 + j + hi * 8;
                const float v = acc[mt][nt][j] + bias;
                const size_t idx = (size_t)m * NA + n;
                if (z == 0)      kexp[idx] = __expf(fminf(fmaxf(v, -60.0f), 30.0f));
                else if (z == 1) vout[idx] = v;
                else             sigr[idx] = 1.0f / (1.0f + __expf(-v));
            }
        }
    }
}

// ---------------------------------------------------------------- cumsum + kv (alpha folded in)
__global__ __launch_bounds__(256)
void k_scan(const float* __restrict__ kexp, const float* __restrict__ vbuf,
            const float* __restrict__ alpha, float* __restrict__ sumk,
            unsigned short* __restrict__ kvb) {
    const int gid = blockIdx.x * blockDim.x + threadIdx.x;  // 0..B*NA-1
    const int b = gid / NA;
    const int n = gid % NA;
    const int h = n / HS;
    float run = 0.0f;
    for (int t = 0; t < T_CTX; ++t) {
        const size_t idx = ((size_t)b * T_CTX + t) * NA + n;
        const float kk = kexp[idx];
        run += kk;
        sumk[idx] = run;
        kvb[idx] = f2bf(alpha[h * T_CTX + t] * kk * vbuf[idx]);
    }
}

// ---------------------------------------------------------------- causal Toeplitz wkv + rwkv epilogue
__global__ __launch_bounds__(128)
void k_wkv(const unsigned short* __restrict__ kvb,
           const unsigned short* __restrict__ twb,   // [H][T] bf16
           const float* __restrict__ beta,           // [H][T]
           const float* __restrict__ sigr,
           const float* __restrict__ sumk,
           unsigned short* __restrict__ rwkvb) {
    __shared__ unsigned short s_tw[T_CTX];
    __shared__ unsigned short s_kv[64][32];   // transposed: [c][u]

    const int tbase = blockIdx.x * 16;
    const int h = blockIdx.y;
    const int b = blockIdx.z;
    const int tid  = threadIdx.x;
    const int lane = tid & 31;
    const int wave = tid >> 5;
    const int l15 = lane & 15;
    const int hi  = lane >> 4;

    for (int i = tid; i < T_CTX; i += 128) s_tw[i] = twb[h * T_CTX + i];
    __syncthreads();

    v8f acc = 0.0f;
    const int t = tbase + l15;          // A-matrix row for this lane
    const int ulimit = tbase + 16;      // u <= tbase+15

    for (int u0 = 0; u0 < ulimit; u0 += 32) {
        // stage kv[u0..u0+31][h*64..h*64+63] transposed into LDS
        {
            const int r  = tid & 31;
            const int c0 = (tid >> 5) * 16;
            const uint4* g = (const uint4*)(kvb + ((size_t)b * T_CTX + u0 + r) * NA + h * HS + c0);
            uint4 q0 = g[0], q1 = g[1];
            const unsigned short* e0 = (const unsigned short*)&q0;
            const unsigned short* e1 = (const unsigned short*)&q1;
            #pragma unroll
            for (int e = 0; e < 8; ++e) s_kv[c0 + e][r] = e0[e];
            #pragma unroll
            for (int e = 0; e < 8; ++e) s_kv[c0 + 8 + e][r] = e1[e];
        }
        __syncthreads();

        FragBF a;
        const bool masked = (u0 + 31 > tbase);
        if (!masked) {
            const int base = (T_CTX - 1) - t + u0 + hi * 8;   // contiguous Toeplitz runs
            #pragma unroll
            for (int s = 0; s < 8; ++s) a.s[s]     = s_tw[base + s];
            #pragma unroll
            for (int s = 0; s < 8; ++s) a.s[8 + s] = s_tw[base + 16 + s];
        } else {
            #pragma unroll
            for (int s = 0; s < 8; ++s) {
                const int u  = u0 + hi * 8 + s;
                const int u2 = u + 16;
                a.s[s]     = (u  <= t) ? s_tw[(T_CTX - 1) - t + u ] : (unsigned short)0;
                a.s[8 + s] = (u2 <= t) ? s_tw[(T_CTX - 1) - t + u2] : (unsigned short)0;
            }
        }

        FragBF bb;
        const int c = wave * 16 + l15;
        bb.q[0] = *(const uint4*)&s_kv[c][hi * 8];
        bb.q[1] = *(const uint4*)&s_kv[c][16 + hi * 8];

        acc = __builtin_amdgcn_wmma_f32_16x16x32_bf16(
            false, a.v, false, bb.v, (short)0, acc, false, false);
        __syncthreads();
    }

    #pragma unroll
    for (int j = 0; j < 8; ++j) {
        const int tt = tbase + j + hi * 8;
        const int n  = h * HS + wave * 16 + l15;
        const size_t idx = ((size_t)b * T_CTX + tt) * NA + n;
        const float wkv = beta[h * T_CTX + tt] * acc[j];
        rwkvb[idx] = f2bf(sigr[idx] * wkv / sumk[idx]);
    }
}

// ---------------------------------------------------------------- output GEMM: out = rwkv @ Wo^T + bo, * gamma[t]
__global__ __launch_bounds__(256)
void k_gemm_out(const unsigned short* __restrict__ A,
                const unsigned short* __restrict__ Wob,
                const float* __restrict__ bo,
                const float* __restrict__ gamma,
                float* __restrict__ out) {
    __shared__ unsigned short sA[2][BM][BK];
    __shared__ unsigned short sB[2][BN][BK];

    const int m0 = blockIdx.y * BM;
    const int n0 = blockIdx.x * BN;
    const int tid  = threadIdx.x;
    const int lane = tid & 31;
    const int wave = tid >> 5;
    const int wm = wave & 1;
    const int wn = wave >> 1;
    const int l15 = lane & 15;
    const int hi  = lane >> 4;

    v8f acc[4][2];
    #pragma unroll
    for (int i = 0; i < 4; ++i)
        #pragma unroll
        for (int j = 0; j < 2; ++j) acc[i][j] = 0.0f;

    const int arow = tid >> 1;
    const int acol = (tid & 1) * 16;

    auto stage = [&](int buf, int k0) {
        const unsigned short* ga = A + (size_t)(m0 + arow) * NA + k0 + acol;
        async_b128(&sA[buf][arow][acol],     ga);
        async_b128(&sA[buf][arow][acol + 8], ga + 8);
        const unsigned short* gb = Wob + (size_t)(n0 + arow) * NA + k0 + acol;
        async_b128(&sB[buf][arow][acol],     gb);
        async_b128(&sB[buf][arow][acol + 8], gb + 8);
    };

    auto compute = [&](int buf) {
        FragBF bfr[2];
        #pragma unroll
        for (int nt = 0; nt < 2; ++nt) {
            const int n = wn * 32 + nt * 16 + l15;
            bfr[nt].q[0] = *(const uint4*)&sB[buf][n][hi * 8];
            bfr[nt].q[1] = *(const uint4*)&sB[buf][n][16 + hi * 8];
        }
        #pragma unroll
        for (int mt = 0; mt < 4; ++mt) {
            FragBF a;
            const int m = wm * 64 + mt * 16 + l15;
            a.q[0] = *(const uint4*)&sA[buf][m][hi * 8];
            a.q[1] = *(const uint4*)&sA[buf][m][16 + hi * 8];
            #pragma unroll
            for (int nt = 0; nt < 2; ++nt)
                acc[mt][nt] = __builtin_amdgcn_wmma_f32_16x16x32_bf16(
                    false, a.v, false, bfr[nt].v, (short)0, acc[mt][nt], false, false);
        }
    };

    const int nk = NA / BK;
    stage(0, 0);
    wait_async0();
    __syncthreads();

    #pragma unroll 1
    for (int i = 0; i < nk - 2; i += 2) {
        stage(1, (i + 1) * BK);
        compute(0);
        wait_async0();
        __syncthreads();
        stage(0, (i + 2) * BK);
        compute(1);
        wait_async0();
        __syncthreads();
    }
    stage(1, (nk - 1) * BK);
    compute(0);
    wait_async0();
    __syncthreads();
    compute(1);

    #pragma unroll
    for (int mt = 0; mt < 4; ++mt) {
        #pragma unroll
        for (int nt = 0; nt < 2; ++nt) {
            const int n = n0 + wn * 32 + nt * 16 + l15;
            const float bias = bo[n];
            #pragma unroll
            for (int j = 0; j < 8; ++j) {
                const int m = m0 + wm * 64 + mt * 16 + j + hi * 8;
                const float g = gamma[m % T_CTX];
                out[(size_t)m * C_EMBD + n] = g * (acc[mt][nt][j] + bias);
            }
        }
    }
}

// ---------------------------------------------------------------- launch
extern "C" void kernel_launch(void* const* d_in, const int* in_sizes, int n_in,
                              void* d_out, int out_size, void* d_ws, size_t ws_size,
                              hipStream_t stream) {
    const float* x     = (const float*)d_in[0];
    const float* tw    = (const float*)d_in[1];
    const float* alpha = (const float*)d_in[2];
    const float* beta  = (const float*)d_in[3];
    const float* gamma = (const float*)d_in[4];
    const float* Wk    = (const float*)d_in[5];
    const float* bk    = (const float*)d_in[6];
    const float* Wv    = (const float*)d_in[7];
    const float* bv    = (const float*)d_in[8];
    const float* Wr    = (const float*)d_in[9];
    const float* br    = (const float*)d_in[10];
    const float* Wo    = (const float*)d_in[11];
    const float* bo    = (const float*)d_in[12];
    float* out = (float*)d_out;

    char* p = (char*)d_ws;
    unsigned short* xs_bf = (unsigned short*)p; p += (size_t)TB * T_CTX * C_EMBD * 2;
    unsigned short* tw_bf = (unsigned short*)p; p += (size_t)NH * T_CTX * 2;
    unsigned short* wk_bf = (unsigned short*)p; p += (size_t)NA * C_EMBD * 2;
    unsigned short* wv_bf = (unsigned short*)p; p += (size_t)NA * C_EMBD * 2;
    unsigned short* wr_bf = (unsigned short*)p; p += (size_t)NA * C_EMBD * 2;
    unsigned short* wo_bf = (unsigned short*)p; p += (size_t)C_EMBD * NA * 2;
    float* kexp  = (float*)p; p += (size_t)TB * T_CTX * NA * 4;
    float* vbuf  = (float*)p; p += (size_t)TB * T_CTX * NA * 4;
    float* sigr  = (float*)p; p += (size_t)TB * T_CTX * NA * 4;
    float* sumk  = (float*)p; p += (size_t)TB * T_CTX * NA * 4;
    unsigned short* kvb   = (unsigned short*)p; p += (size_t)TB * T_CTX * NA * 2;
    unsigned short* rwkvb = (unsigned short*)p; p += (size_t)TB * T_CTX * NA * 2;

    const int totalX = TB * T_CTX * C_EMBD;
    k_shift_cast<<<totalX / 256, 256, 0, stream>>>(x, xs_bf, totalX);
    k_cast<<<(NA * C_EMBD) / 256, 256, 0, stream>>>(Wk, wk_bf, NA * C_EMBD);
    k_cast<<<(NA * C_EMBD) / 256, 256, 0, stream>>>(Wv, wv_bf, NA * C_EMBD);
    k_cast<<<(NA * C_EMBD) / 256, 256, 0, stream>>>(Wr, wr_bf, NA * C_EMBD);
    k_cast<<<(C_EMBD * NA) / 256, 256, 0, stream>>>(Wo, wo_bf, C_EMBD * NA);
    k_cast<<<(NH * T_CTX) / 256, 256, 0, stream>>>(tw, tw_bf, NH * T_CTX);

    k_gemm_kvr<<<dim3(NA / BN, (TB * T_CTX) / BM, 3), 256, 0, stream>>>(
        xs_bf, wk_bf, wv_bf, wr_bf, bk, bv, br, kexp, vbuf, sigr);

    k_scan<<<(TB * NA) / 256, 256, 0, stream>>>(kexp, vbuf, alpha, sumk, kvb);

    k_wkv<<<dim3(T_CTX / 16, NH, TB), 128, 0, stream>>>(
        kvb, tw_bf, beta, sigr, sumk, rwkvb);

    k_gemm_out<<<dim3(C_EMBD / BN, (TB * T_CTX) / BM, 1), 256, 0, stream>>>(
        rwkvb, wo_bf, bo, gamma, out);
}
